// GNN_node_21930103014155
// MI455X (gfx1250) — matrix operand order repaired and checked by reading, not compiled, not run on hardware
//
#include <hip/hip_runtime.h>

// ---------------- problem constants (match reference) ----------------
#define NN   50000
#define EE   300000
#define DD   256
#define LL   5
#define MPAD 50048          // 391 * 128, row-padded for full-EXEC WMMA tiles
#define CAP  64             // max neighbors stored per node (Poisson(6) -> safe)
#define BN_EPS_C 1e-5f

// LDS weight tile: 256 rows x (256+8) bf16 -> +16B row pad, bank-conflict-free
#define BPAD      264
#define LDS_WB    (256 * BPAD * 2)          // 135168 B
#define LDS_TOTAL (LDS_WB + 2 * DD * 4)     // + 512 floats of BN partials

// ---------------- vector types for WMMA ----------------
typedef __attribute__((ext_vector_type(16))) __bf16        v16bf;
typedef __attribute__((ext_vector_type(8)))  float         v8f;
typedef __attribute__((ext_vector_type(4)))  unsigned int  v4u;
typedef __attribute__((ext_vector_type(8)))  unsigned int  v8u;

__device__ __forceinline__ unsigned short f2bf(float f) {
    unsigned int u = __float_as_uint(f);
    unsigned int r = (u + 0x7FFFu + ((u >> 16) & 1u)) >> 16;
    return (unsigned short)r;
}

// A-fragment: two 16B chunks (K = base..base+7 and base+16..base+23)
__device__ __forceinline__ v16bf load_frag_split(const unsigned short* p0,
                                                 const unsigned short* p1) {
    v4u lo = *reinterpret_cast<const v4u*>(p0);
    v4u hi = *reinterpret_cast<const v4u*>(p1);
    v8u t;
    t[0]=lo[0]; t[1]=lo[1]; t[2]=lo[2]; t[3]=lo[3];
    t[4]=hi[0]; t[5]=hi[1]; t[6]=hi[2]; t[7]=hi[3];
    return __builtin_bit_cast(v16bf, t);
}

// B-fragment: 16 consecutive bf16 (32B) per lane
__device__ __forceinline__ v16bf load_frag_contig(const unsigned short* p) {
    v8u t = *reinterpret_cast<const v8u*>(p);
    return __builtin_bit_cast(v16bf, t);
}

// ---------------- weight convert: WbT[l][j][k] = bf16(W[l][k][j]) ----------------
__global__ __launch_bounds__(256) void k_wconv(const float* __restrict__ W,
                                               unsigned short* __restrict__ Wb) {
    int gid = blockIdx.x * 256 + threadIdx.x;          // l*65536 + j*256 + k
    if (gid >= LL * DD * DD) return;
    int k = gid & 255;
    int j = (gid >> 8) & 255;
    int l = gid >> 16;
    Wb[gid] = f2bf(W[(size_t)l * DD * DD + (size_t)k * DD + j]);
}

// ---------------- zero helpers ----------------
__global__ void k_zero_f32(float* __restrict__ p, int n) {
    int gid = blockIdx.x * 256 + threadIdx.x;
    if (gid < n) p[gid] = 0.f;
}
__global__ void k_zero_i32(int* __restrict__ p, int n) {
    int gid = blockIdx.x * 256 + threadIdx.x;
    if (gid < n) p[gid] = 0;
}
// zero bf16 pad rows NN..MPAD-1 once per launch
__global__ void k_zero_tail(unsigned short* __restrict__ zb) {
    int gid = blockIdx.x * 256 + threadIdx.x;
    if (gid < (MPAD - NN) * DD) zb[(size_t)NN * DD + gid] = 0;
}

// ---------------- bucket build (once per launch; int atomics only) ----------------
__global__ __launch_bounds__(256) void k_fill(const int* __restrict__ src,
                                              const int* __restrict__ dst,
                                              int* __restrict__ cnt,
                                              int* __restrict__ bucket) {
    int e = blockIdx.x * 256 + threadIdx.x;
    if (e >= EE) return;
    int d = dst[e];
    int pos = atomicAdd(&cnt[d], 1);
    if (pos < CAP) bucket[(size_t)d * CAP + pos] = src[e];
}

// ---------------- GIN aggregate (no float atomics): one block per node ----------------
// zb[n][d] = bf16( (1+eps)*h[n][d] + sum_nbr relu(h[nbr][d]) )
__global__ __launch_bounds__(256) void k_aggregate(const float* __restrict__ h,
                                                   const int* __restrict__ cnt,
                                                   const int* __restrict__ bucket,
                                                   const float* __restrict__ epsp,
                                                   unsigned short* __restrict__ zb) {
    int n = blockIdx.x;
    int d = threadIdx.x;
    float acc = (1.0f + epsp[0]) * h[(size_t)n * DD + d];
    int deg = cnt[n];
    if (deg > CAP) deg = CAP;
    for (int i = 0; i < deg; ++i) {
        int s = bucket[(size_t)n * CAP + i];
        float v = h[(size_t)s * DD + d];
        acc += v > 0.f ? v : 0.f;
    }
    zb[(size_t)n * DD + d] = f2bf(acc);
}

// ---------------- WMMA GEMM + fused BN partial stats ----------------
// C[NN x 256] = A[MPAD x 256](bf16) @ B + bias ; stats += per-feature {sum, sumsq}
// BT is [256(out) x 256(k)] bf16, staged in LDS (row stride BPAD, conflict-free).
__global__ __launch_bounds__(256) void k_gemm(const unsigned short* __restrict__ A,
                                              const unsigned short* __restrict__ BT,
                                              const float* __restrict__ bias,
                                              float* __restrict__ C,
                                              float* __restrict__ stats) {
    extern __shared__ unsigned short smem_us[];
    unsigned short* bsm = smem_us;                              // 256 x BPAD bf16
    float* ssm = reinterpret_cast<float*>(smem_us + 256 * BPAD); // 512 f32

    // ---- stage weight tile into LDS (16B chunks, coalesced) + zero stat partials
    for (int i = threadIdx.x; i < 256 * 32; i += 256) {
        int row = i >> 5;
        int off = (i & 31) << 3;                                // 8 ushorts / chunk
        v4u d = *reinterpret_cast<const v4u*>(BT + (size_t)row * DD + off);
        *reinterpret_cast<v4u*>(bsm + row * BPAD + off) = d;
    }
    ssm[threadIdx.x] = 0.f;
    ssm[DD + threadIdx.x] = 0.f;
    __syncthreads();

    const int lane = threadIdx.x & 31;
    const int wave = threadIdx.x >> 5;
    const int rowBase = blockIdx.x * 128 + wave * 16;
    const int m  = lane & 15;          // A row / B col / C col within tile
    const int g  = lane >> 4;          // lane group (K packing select)
    const int kbA = g * 8;
    const int kbB = g * 8;

    v8f acc[16];
#pragma unroll
    for (int j = 0; j < 16; ++j) {
        float bv = bias[j * 16 + m];
#pragma unroll
        for (int r = 0; r < 8; ++r) acc[j][r] = bv;
    }

    const unsigned short* arow = A + (size_t)(rowBase + m) * DD;
#pragma unroll
    for (int kk = 0; kk < 8; ++kk) {
        v16bf a = load_frag_split(arow + kk * 32 + kbA,
                                  arow + kk * 32 + kbA + 16);
#pragma unroll
        for (int j = 0; j < 16; ++j) {
            const unsigned short* bp = bsm + (j * 16 + m) * BPAD + kk * 32 + kbB;
            v16bf b = load_frag_split(bp, bp + 16);
            acc[j] = __builtin_amdgcn_wmma_f32_16x16x32_bf16(
                false, a, false, b, (short)0, acc[j], false, false);
        }
    }

    // ---- epilogue: store C + accumulate BN partials into LDS (ds_add_f32)
    const int rowoff = g * 8;
#pragma unroll
    for (int j = 0; j < 16; ++j) {
        int col = j * 16 + m;
        float s = 0.f, s2 = 0.f;
#pragma unroll
        for (int r = 0; r < 8; ++r) {
            int row = rowBase + rowoff + r;
            if (row < NN) {
                float v = acc[j][r];
                C[(size_t)row * DD + col] = v;
                s += v;
                s2 += v * v;
            }
        }
        atomicAdd(&ssm[col], s);
        atomicAdd(&ssm[DD + col], s2);
    }
    __syncthreads();
    for (int i = threadIdx.x; i < 2 * DD; i += 256)
        atomicAdd(&stats[i], ssm[i]);
}

// ---------------- BN: finalize scale/shift ----------------
__global__ void k_bnfinalize(const float* __restrict__ stats,
                             const float* __restrict__ gamma,
                             const float* __restrict__ beta,
                             float* __restrict__ ss) {
    int d = threadIdx.x;
    float inv_n = 1.0f / (float)NN;
    float mean = stats[d] * inv_n;
    float var  = stats[DD + d] * inv_n - mean * mean;
    float rs = rsqrtf(var + BN_EPS_C);
    float sc = gamma[d] * rs;
    ss[d] = sc;
    ss[DD + d] = beta[d] - mean * sc;
}

// ---------------- BN apply + ReLU + bf16 convert (NN rows) ----------------
__global__ __launch_bounds__(256) void k_bnrelu_bf16(const float* __restrict__ t,
                                                     const float* __restrict__ ss,
                                                     unsigned short* __restrict__ zb) {
    int gid = blockIdx.x * 256 + threadIdx.x;          // vec4 over NN*DD
    if (gid >= NN * DD / 4) return;
    int idx = gid << 2;
    int d0 = idx & 255;
    float4 v = reinterpret_cast<const float4*>(t)[gid];
    float o0 = v.x * ss[d0 + 0] + ss[DD + d0 + 0];
    float o1 = v.y * ss[d0 + 1] + ss[DD + d0 + 1];
    float o2 = v.z * ss[d0 + 2] + ss[DD + d0 + 2];
    float o3 = v.w * ss[d0 + 3] + ss[DD + d0 + 3];
    zb[idx + 0] = f2bf(o0 > 0.f ? o0 : 0.f);
    zb[idx + 1] = f2bf(o1 > 0.f ? o1 : 0.f);
    zb[idx + 2] = f2bf(o2 > 0.f ? o2 : 0.f);
    zb[idx + 3] = f2bf(o3 > 0.f ? o3 : 0.f);
}

// ---------------- BN apply (+optional ReLU) -> f32 output ----------------
__global__ __launch_bounds__(256) void k_bnfinal(const float* __restrict__ t,
                                                 const float* __restrict__ ss,
                                                 float* __restrict__ out,
                                                 int dorelu) {
    int gid = blockIdx.x * 256 + threadIdx.x;          // vec4 over NN*DD
    if (gid >= NN * DD / 4) return;
    int idx = gid << 2;
    int d0 = idx & 255;
    float4 v = reinterpret_cast<const float4*>(t)[gid];
    float o0 = v.x * ss[d0 + 0] + ss[DD + d0 + 0];
    float o1 = v.y * ss[d0 + 1] + ss[DD + d0 + 1];
    float o2 = v.z * ss[d0 + 2] + ss[DD + d0 + 2];
    float o3 = v.w * ss[d0 + 3] + ss[DD + d0 + 3];
    if (dorelu) {
        o0 = o0 > 0.f ? o0 : 0.f; o1 = o1 > 0.f ? o1 : 0.f;
        o2 = o2 > 0.f ? o2 : 0.f; o3 = o3 > 0.f ? o3 : 0.f;
    }
    float4 o; o.x = o0; o.y = o1; o.z = o2; o.w = o3;
    reinterpret_cast<float4*>(out)[gid] = o;
}

// ---------------- host orchestration ----------------
extern "C" void kernel_launch(void* const* d_in, const int* in_sizes, int n_in,
                              void* d_out, int out_size, void* d_ws, size_t ws_size,
                              hipStream_t stream) {
    (void)in_sizes; (void)n_in; (void)out_size; (void)ws_size;
    const float* x    = (const float*)d_in[0];
    const int*   src  = (const int*)d_in[1];
    const int*   dst  = (const int*)d_in[2];
    const float* W1   = (const float*)d_in[3];
    const float* b1   = (const float*)d_in[4];
    const float* g1   = (const float*)d_in[5];
    const float* bt1  = (const float*)d_in[6];
    const float* W2   = (const float*)d_in[7];
    const float* b2   = (const float*)d_in[8];
    const float* eps  = (const float*)d_in[9];
    const float* bn_g = (const float*)d_in[10];
    const float* bn_b = (const float*)d_in[11];

    char* ws = (char*)d_ws;
    const size_t NB = (size_t)NN * DD * sizeof(float);          // 51.2 MB
    float* hA = (float*)ws;                      ws += NB;
    float* hB = (float*)ws;                      ws += NB;
    float* zf = (float*)ws;                      ws += NB;       // t1 / t2
    unsigned short* zb  = (unsigned short*)ws;   ws += (size_t)MPAD * DD * 2;
    unsigned short* wb1 = (unsigned short*)ws;   ws += (size_t)LL * DD * DD * 2;
    unsigned short* wb2 = (unsigned short*)ws;   ws += (size_t)LL * DD * DD * 2;
    float* stats = (float*)ws;                   ws += 2 * DD * sizeof(float);
    float* ssbuf = (float*)ws;                   ws += 2 * DD * sizeof(float);
    int* cnt     = (int*)ws;                     ws += (size_t)NN * sizeof(int);
    int* bucket  = (int*)ws;                     ws += (size_t)NN * CAP * sizeof(int);

    const int blkW    = (LL * DD * DD + 255) / 256;   // 1280
    const int blkVec4 = (NN * DD / 4) / 256;          // 12500
    const int blkEdge = (EE + 255) / 256;             // 1172
    const int blkGemm = MPAD / 128;                   // 391

    // once per launch: bf16 weights, zb pad rows, edge buckets
    k_wconv<<<blkW, 256, 0, stream>>>(W1, wb1);
    k_wconv<<<blkW, 256, 0, stream>>>(W2, wb2);
    k_zero_tail<<<(MPAD - NN) * DD / 256, 256, 0, stream>>>(zb);
    k_zero_i32<<<(NN + 255) / 256, 256, 0, stream>>>(cnt, NN);
    k_fill<<<blkEdge, 256, 0, stream>>>(src, dst, cnt, bucket);

    const float* hin = x;
    for (int l = 0; l < LL; ++l) {
        // GIN aggregate -> bf16 A matrix (no float atomics)
        k_aggregate<<<NN, 256, 0, stream>>>(hin, cnt, bucket, eps + l, zb);
        // Linear 1 (+bias) via WMMA, fused BN stats
        k_zero_f32<<<2, 256, 0, stream>>>(stats, 2 * DD);
        k_gemm<<<blkGemm, 256, LDS_TOTAL, stream>>>(zb, wb1 + (size_t)l * DD * DD,
                                                    b1 + (size_t)l * DD, zf, stats);
        k_bnfinalize<<<1, 256, 0, stream>>>(stats, g1 + (size_t)l * DD,
                                            bt1 + (size_t)l * DD, ssbuf);
        k_bnrelu_bf16<<<blkVec4, 256, 0, stream>>>(zf, ssbuf, zb);
        // Linear 2 (+bias) via WMMA, fused BN stats
        k_zero_f32<<<2, 256, 0, stream>>>(stats, 2 * DD);
        k_gemm<<<blkGemm, 256, LDS_TOTAL, stream>>>(zb, wb2 + (size_t)l * DD * DD,
                                                    b2 + (size_t)l * DD, zf, stats);
        k_bnfinalize<<<1, 256, 0, stream>>>(stats, bn_g + (size_t)l * DD,
                                            bn_b + (size_t)l * DD, ssbuf);
        float* hout = (l == LL - 1) ? (float*)d_out : ((l & 1) ? hB : hA);
        k_bnfinal<<<blkVec4, 256, 0, stream>>>(zf, ssbuf, hout, (l < LL - 1) ? 1 : 0);
        hin = hout;
    }
}
